// GNN_Layer_20547123544613
// MI455X (gfx1250) — compile-verified, matrix operand order checked by reading, not run on hardware
//
#include <hip/hip_runtime.h>
#include <stdint.h>

// Problem constants (match reference)
#define NN    50000
#define VF    128
#define KN    10
#define TILEM 16

typedef __attribute__((ext_vector_type(16))) __bf16       v16bf;
typedef __attribute__((ext_vector_type(8)))  float        v8f;
typedef __attribute__((ext_vector_type(8)))  unsigned int v8u;

// Split an f32 into bf16 hi (truncation) + bf16 lo (truncated residual).
// Combined representation error ~2^-16 relative; f32 WMMA accumulation.
__device__ __forceinline__ void bfsplit(float x, unsigned& h, unsigned& l) {
    unsigned u     = __float_as_uint(x);
    unsigned hbits = u & 0xffff0000u;
    h = hbits >> 16;
    float r = x - __uint_as_float(hbits);
    l = __float_as_uint(r) >> 16;
}

// Pack 4 floats into (hi0|hi1, hi2|hi3) and (lo0|lo1, lo2|lo3) u32 pairs.
__device__ __forceinline__ void pack4(float4 v, uint2& hi, uint2& lo) {
    unsigned h0, l0, h1, l1, h2, l2, h3, l3;
    bfsplit(v.x, h0, l0);
    bfsplit(v.y, h1, l1);
    bfsplit(v.z, h2, l2);
    bfsplit(v.w, h3, l3);
    hi = make_uint2(h0 | (h1 << 16), h2 | (h3 << 16));
    lo = make_uint2(l0 | (l1 << 16), l2 | (l3 << 16));
}

// ---------------------------------------------------------------------------
// Pack W matrices (Wsv, Wsr, Wdr) into WMMA B-fragment layout, hi+lo planes.
// Layout (u32 units): idx = ((((term*3 + m)*8 + ntile)*4 + kchunk)*32 + lane)*8 + j
//   B is K(32) x N(16) bf16; lane<16 -> N=lane, K in {0..7,16..23}+32*c
//   lane>=16 -> N=lane-16, K offsets +8. VGPR j holds K pair (2 bf16 per u32).
// Total: 2*3*8*4*32*8 u32 = 192 KB in d_ws.
// ---------------------------------------------------------------------------
__global__ void pack_weights(const float* __restrict__ Wsv,
                             const float* __restrict__ Wdr,
                             const float* __restrict__ Wsr,
                             unsigned int* __restrict__ bpack) {
    int idx = blockIdx.x * blockDim.x + threadIdx.x;   // 0 .. 49151
    if (idx >= 2 * 3 * 8 * 4 * 32 * 8) return;
    int j    = idx & 7;
    int lane = (idx >> 3) & 31;
    int c    = (idx >> 8) & 3;
    int nt   = (idx >> 10) & 7;
    int m    = (idx >> 13) % 3;
    int term = idx / (3 << 13);                        // 0 = hi plane, 1 = lo plane
    const float* W = (m == 0) ? Wsv : (m == 1) ? Wsr : Wdr;
    int n  = nt * 16 + (lane & 15);
    int k0 = c * 32 + ((j >> 2) << 4) + ((j & 3) << 1) + (((lane >> 4) & 1) << 3);
    float x0 = W[(size_t)k0 * VF + n];
    float x1 = W[(size_t)(k0 + 1) * VF + n];
    unsigned h0, l0, h1, l1;
    bfsplit(x0, h0, l0);
    bfsplit(x1, h1, l1);
    bpack[idx] = term ? (l0 | (l1 << 16)) : (h0 | (h1 << 16));
}

// ---------------------------------------------------------------------------
// Fused gather-mean + 3-matrix GEMM + ReLU.
// Block = 256 threads (8 waves), owns 16 nodes. Gather phase converts A to
// packed bf16 hi/lo ONCE into LDS; WMMA phase is pure ds_load_b128 + wmma.
// Wave w computes output columns [16w, 16w+16) for all 16 nodes.
// ---------------------------------------------------------------------------
__global__ __launch_bounds__(256)
void gnn_fused(const float* __restrict__ Z0, const int* __restrict__ sn0,
               const int* __restrict__ dn0,
               const float* __restrict__ Z1, const int* __restrict__ sn1,
               const int* __restrict__ dn1,
               const unsigned int* __restrict__ bpack,
               float* __restrict__ out) {
    // A planes in bf16, row-major: [matrix][row][VF] bf16 == [..][..][VF/2] u32.
    __shared__ unsigned int Ahi[3][TILEM][VF / 2];   // 12 KB
    __shared__ unsigned int Alo[3][TILEM][VF / 2];   // 12 KB

    const int   p  = blockIdx.y;
    const float* Z = p ? Z1 : Z0;
    const int*  sn = p ? sn1 : sn0;
    const int*  dn = p ? dn1 : dn0;
    float* r = out + (size_t)p * 7400000;   // r_p slot in concatenated output

    const int tile = blockIdx.x;
    const int w    = threadIdx.x >> 5;
    const int lane = threadIdx.x & 31;

    // ---- Gather phase: each wave fills 2 node rows. Lane owns 4 columns. ----
    for (int s = 0; s < 2; ++s) {
        const int m = w * 2 + s;
        const int g = tile * TILEM + m;

        // A0: the node's own Z row -> split+pack once
        {
            float4 z = *(const float4*)(Z + (size_t)g * VF + lane * 4);
            uint2 hi, lo;
            pack4(z, hi, lo);
            *(uint2*)&Ahi[0][m][lane * 2] = hi;
            *(uint2*)&Alo[0][m][lane * 2] = lo;
        }

        // masked mean over neighbor lists -> split+pack once
        #pragma unroll
        for (int list = 0; list < 2; ++list) {
            const int* nb = list ? dn : sn;
            int my = (lane < KN) ? nb[g * KN + lane] : -1;
            float4 acc = make_float4(0.f, 0.f, 0.f, 0.f);
            int cnt = 0;
            #pragma unroll
            for (int k = 0; k < KN; ++k) {
                int id = __shfl(my, k);
                if (id > -1) {
                    const float4 zz = *(const float4*)(Z + (size_t)id * VF + lane * 4);
                    acc.x += zz.x; acc.y += zz.y; acc.z += zz.z; acc.w += zz.w;
                    ++cnt;
                }
            }
            const float sc = 1.0f / (float)(cnt > 0 ? cnt : 1);
            float4 o = make_float4(acc.x * sc, acc.y * sc, acc.z * sc, acc.w * sc);
            uint2 hi, lo;
            pack4(o, hi, lo);
            *(uint2*)&Ahi[1 + list][m][lane * 2] = hi;
            *(uint2*)&Alo[1 + list][m][lane * 2] = lo;
        }
    }
    __syncthreads();

    // ---- WMMA phase: wave w -> n-tile w. 4 k-chunks x 3 matrices x 3 terms. ----
    const int nt = w;
    v8f acc = {0.f, 0.f, 0.f, 0.f, 0.f, 0.f, 0.f, 0.f};
    const v16bf* B = (const v16bf*)bpack;

    // Per-lane row base inside a matrix plane (u32 units): row = lane&15.
    const int rowbase = (lane & 15) * (VF / 2);

    #pragma unroll
    for (int c = 0; c < 4; ++c) {
        // u32 offset of this lane's first K element: (c*32 + 8*(lane>=16)) / 2
        const int koff = c * 16 + ((lane >> 4) << 2);
        #pragma unroll
        for (int m = 0; m < 3; ++m) {
            const unsigned int* hp = &Ahi[m][0][0] + rowbase + koff;
            const unsigned int* lp = &Alo[m][0][0] + rowbase + koff;
            // Fragment = K 0..7 (4 u32) then K 16..23 (4 u32, +8 u32 offset)
            uint4 h0 = *(const uint4*)(hp);
            uint4 h1 = *(const uint4*)(hp + 8);
            uint4 l0 = *(const uint4*)(lp);
            uint4 l1 = *(const uint4*)(lp + 8);
            v8u phv = {h0.x, h0.y, h0.z, h0.w, h1.x, h1.y, h1.z, h1.w};
            v8u plv = {l0.x, l0.y, l0.z, l0.w, l1.x, l1.y, l1.z, l1.w};
            v16bf ah = __builtin_bit_cast(v16bf, phv);
            v16bf al = __builtin_bit_cast(v16bf, plv);

            const int fh = ((0 * 3 + m) * 8 + nt) * 4 + c;
            const int fl = ((1 * 3 + m) * 8 + nt) * 4 + c;
            v16bf bh = B[fh * 32 + lane];
            v16bf bl = B[fl * 32 + lane];

            acc = __builtin_amdgcn_wmma_f32_16x16x32_bf16(false, ah, false, bh,
                                                          (short)0, acc, false, false);
            acc = __builtin_amdgcn_wmma_f32_16x16x32_bf16(false, al, false, bh,
                                                          (short)0, acc, false, false);
            acc = __builtin_amdgcn_wmma_f32_16x16x32_bf16(false, ah, false, bl,
                                                          (short)0, acc, false, false);
        }
    }

    // ---- ReLU + store. D layout: VGPR rr -> M=rr (lanes 0-15), M=rr+8 (16-31). ----
    const int col    = nt * 16 + (lane & 15);
    const int rowoff = (lane >> 4) * 8;
    #pragma unroll
    for (int rr = 0; rr < 8; ++rr) {
        const int row = tile * TILEM + rowoff + rr;
        float v = acc[rr];
        r[(size_t)row * VF + col] = v > 0.f ? v : 0.f;
    }
}

// ---------------------------------------------------------------------------
extern "C" void kernel_launch(void* const* d_in, const int* in_sizes, int n_in,
                              void* d_out, int out_size, void* d_ws, size_t ws_size,
                              hipStream_t stream) {
    (void)in_sizes; (void)n_in; (void)out_size; (void)ws_size;

    const float* Z0  = (const float*)d_in[0];
    const int*   sn0 = (const int*)d_in[1];
    const int*   dn0 = (const int*)d_in[2];
    const float* Z1  = (const float*)d_in[3];
    const int*   sn1 = (const int*)d_in[4];
    const int*   dn1 = (const int*)d_in[5];
    const float* Wsv = (const float*)d_in[6];
    const float* Wdr = (const float*)d_in[7];
    const float* Wsr = (const float*)d_in[8];

    unsigned int* bpack = (unsigned int*)d_ws;   // 192 KB used
    float* out = (float*)d_out;

    // 1) Pack weights into WMMA B-fragment layout (hi/lo bf16 planes)
    pack_weights<<<192, 256, 0, stream>>>(Wsv, Wdr, Wsr, bpack);

    // 2) Fused gather + GEMM + ReLU for both proteins
    dim3 grid(NN / TILEM, 2);
    gnn_fused<<<grid, 256, 0, stream>>>(Z0, sn0, dn0, Z1, sn1, dn1, bpack, out);

    // 3) Neighbor index pass-through (bitwise into the float-typed output slots)
    hipMemcpyAsync(out + 6400000,  sn0, (size_t)NN * KN * 4, hipMemcpyDeviceToDevice, stream);
    hipMemcpyAsync(out + 6900000,  dn0, (size_t)NN * KN * 4, hipMemcpyDeviceToDevice, stream);
    hipMemcpyAsync(out + 13800000, sn1, (size_t)NN * KN * 4, hipMemcpyDeviceToDevice, stream);
    hipMemcpyAsync(out + 14300000, dn1, (size_t)NN * KN * 4, hipMemcpyDeviceToDevice, stream);
}
